// GNNStack_3539053052342
// MI455X (gfx1250) — matrix operand order repaired and checked by reading, not compiled, not run on hardware
//
#include <hip/hip_runtime.h>

// Problem constants (from the reference)
#define N_NODES 100000
#define G_NUM   128
#define DIN     512
#define DH      1024
#define DOUT    512
#define TILE    32        // nodes per workgroup; 100000 / 32 = 3125 exactly
#define THREADS 128       // 4 waves (wave32)

typedef __attribute__((ext_vector_type(16))) __bf16 v16bf;
typedef __attribute__((ext_vector_type(8)))  __bf16 v8bf;
typedef __attribute__((ext_vector_type(8)))  float  v8f;

__device__ __forceinline__ __bf16 f2bf(float f) {
  // round-to-nearest-even fp32 -> bf16
  unsigned u = __builtin_bit_cast(unsigned, f);
  unsigned r = u + 0x7FFFu + ((u >> 16) & 1u);
  unsigned short h = (unsigned short)(r >> 16);
  return __builtin_bit_cast(__bf16, h);
}

// A fragment (16x32 bf16) from row-major LDS tile.
// ISA layout: lane l holds row M=l%16; K chunks [ (l/16)*8 .. +7 ] and [ 16+(l/16)*8 .. +7 ].
__device__ __forceinline__ v16bf load_a_lds(const __bf16* base, int rowStride,
                                            int m0, int k0, int lane) {
  const int m  = m0 + (lane & 15);
  const int hk = (lane >> 4) * 8;
  const __bf16* p = base + m * rowStride + k0 + hk;
  union { v16bf v; v8bf h[2]; } u;
  u.h[0] = *(const v8bf*)(p);
  u.h[1] = *(const v8bf*)(p + 16);
  return u.v;
}

// B fragment (32x16 bf16) from pre-transposed weights Wt[N][K] (row-major).
// ISA layout: lane l holds column N=l%16; K chunks [ (l/16)*8 .. +7 ] and [ 16+(l/16)*8 .. +7 ].
__device__ __forceinline__ v16bf load_b_glb(const __bf16* __restrict__ wt, int K,
                                            int n0, int k0, int lane) {
  const int n  = n0 + (lane & 15);
  const int kk = k0 + (lane >> 4) * 8;
  const __bf16* p = wt + (size_t)n * K + kk;
  union { v16bf v; v8bf h[2]; } u;
  u.h[0] = *(const v8bf*)(p);
  u.h[1] = *(const v8bf*)(p + 16);
  return u.v;
}

__device__ __forceinline__ v8f wmma_bf16(v16bf a, v16bf b, v8f c) {
  return __builtin_amdgcn_wmma_f32_16x16x32_bf16(
      /*neg_a=*/false, a, /*neg_b=*/false, b,
      /*c_mod=*/(short)0, c, /*reuse_a=*/false, /*reuse_b=*/false);
}

// One fused layer: dst = relu(src @ W + b), src/dst row-major bf16 in LDS.
// Each wave handles both 16-row strips for its column range (B reused for 2 WMMAs).
template<int K>
__device__ __forceinline__ void mlp_layer(const __bf16* src, int srcStride,
                                          const __bf16* __restrict__ wt,
                                          const float* __restrict__ bias,
                                          __bf16* dst, int dstStride,
                                          int ncol0, int ncols, int lane) {
  for (int nt = 0; nt < ncols; nt += 16) {
    const int n0 = ncol0 + nt;
    v8f acc0 = {}; v8f acc1 = {};
    #pragma unroll 4
    for (int k0 = 0; k0 < K; k0 += 32) {
      const v16bf b  = load_b_glb(wt, K, n0, k0, lane);
      const v16bf a0 = load_a_lds(src, srcStride, 0,  k0, lane);
      const v16bf a1 = load_a_lds(src, srcStride, 16, k0, lane);
      acc0 = wmma_bf16(a0, b, acc0);
      acc1 = wmma_bf16(a1, b, acc1);
    }
    const int col = n0 + (lane & 15);
    const float bv = bias[col];
    const int mb0 = (lane >> 4) * 8;      // C/D layout: row = r + 8*(lane/16)
    const int mb1 = 16 + mb0;
    #pragma unroll
    for (int r = 0; r < 8; ++r) {
      float t0 = acc0[r] + bv; t0 = t0 > 0.f ? t0 : 0.f;
      float t1 = acc1[r] + bv; t1 = t1 > 0.f ? t1 : 0.f;
      dst[(mb0 + r) * dstStride + col] = f2bf(t0);
      dst[(mb1 + r) * dstStride + col] = f2bf(t1);
    }
  }
}

// ---- Kernel 0: convert+transpose weights to bf16 Wt[N][K]; zero pool sums ----
__global__ void __launch_bounds__(256) prep_kernel(
    const float* __restrict__ W0, const float* __restrict__ W1,
    const float* __restrict__ W2,
    __bf16* __restrict__ Wt0, __bf16* __restrict__ Wt1, __bf16* __restrict__ Wt2,
    float* __restrict__ sums) {
  const int i = blockIdx.x * 256 + threadIdx.x;
  if (i < DIN * DH)   { int k = i / DH;   int n = i % DH;   Wt0[(size_t)n * DIN + k] = f2bf(W0[i]); }
  if (i < DH * DH)    { int k = i / DH;   int n = i % DH;   Wt1[(size_t)n * DH  + k] = f2bf(W1[i]); }
  if (i < DH * DOUT)  { int k = i / DOUT; int n = i % DOUT; Wt2[(size_t)n * DH  + k] = f2bf(W2[i]); }
  if (i < G_NUM * DOUT) sums[i] = 0.f;
}

// ---- Kernel 1: fused 3-layer MLP (WMMA) + per-graph sum pooling ----
__global__ void __launch_bounds__(THREADS) fused_mlp_pool(
    const float* __restrict__ x, const long long* __restrict__ batch,
    const __bf16* __restrict__ Wt0, const __bf16* __restrict__ Wt1,
    const __bf16* __restrict__ Wt2,
    const float* __restrict__ b0, const float* __restrict__ b1,
    const float* __restrict__ b2,
    float* __restrict__ sums) {
  extern __shared__ __bf16 lds[];
  __bf16* X  = lds;                  // TILE x DIN   (32 KB)
  __bf16* H1 = X  + TILE * DIN;      // TILE x DH    (64 KB)
  __bf16* H2 = H1 + TILE * DH;       // TILE x DH    (64 KB)
  const int tid  = threadIdx.x;
  const int lane = tid & 31;
  const int wave = tid >> 5;         // 0..3, owns a column quarter
  const int tb   = blockIdx.x * TILE;

  // Stage x tile -> LDS as bf16 (coalesced float4 reads)
  {
    const float* xg = x + (size_t)tb * DIN;
    for (int i = tid * 4; i < TILE * DIN; i += THREADS * 4) {
      const float4 v = *(const float4*)(xg + i);
      X[i + 0] = f2bf(v.x); X[i + 1] = f2bf(v.y);
      X[i + 2] = f2bf(v.z); X[i + 3] = f2bf(v.w);
    }
  }
  __syncthreads();

  mlp_layer<DIN>(X,  DIN, Wt0, b0, H1, DH, wave * (DH / 4), DH / 4, lane);
  __syncthreads();
  mlp_layer<DH >(H1, DH,  Wt1, b1, H2, DH, wave * (DH / 4), DH / 4, lane);
  __syncthreads();

  // Layer 2 + pooling. batch is sorted, so a tile is usually one graph.
  const long long gFirst = batch[tb];
  const long long gS0l   = batch[tb + 15];
  const long long gS1f   = batch[tb + 16];
  const long long gLast  = batch[tb + 31];
  const int ncol0 = wave * (DOUT / 4);

  for (int nt = 0; nt < DOUT / 4; nt += 16) {
    const int n0 = ncol0 + nt;
    v8f acc0 = {}; v8f acc1 = {};
    #pragma unroll 4
    for (int k0 = 0; k0 < DH; k0 += 32) {
      const v16bf b  = load_b_glb(Wt2, DH, n0, k0, lane);
      const v16bf a0 = load_a_lds(H2, DH, 0,  k0, lane);
      const v16bf a1 = load_a_lds(H2, DH, 16, k0, lane);
      acc0 = wmma_bf16(a0, b, acc0);
      acc1 = wmma_bf16(a1, b, acc1);
    }
    const int col = n0 + (lane & 15);
    const float bv = b2[col];
    float v0[8], v1[8];
    #pragma unroll
    for (int r = 0; r < 8; ++r) {
      float t0 = acc0[r] + bv; v0[r] = t0 > 0.f ? t0 : 0.f;
      float t1 = acc1[r] + bv; v1[r] = t1 > 0.f ? t1 : 0.f;
    }

    if (gFirst == gLast) {
      // whole tile belongs to one graph: in-register reduce + 1 atomic/col
      float s = 0.f;
      #pragma unroll
      for (int r = 0; r < 8; ++r) s += v0[r] + v1[r];
      s += __shfl_xor(s, 16, 32);          // combine row halves (wave32)
      if (lane < 16) atomicAdd(&sums[(int)gFirst * DOUT + col], s);
    } else {
      // strip 0
      if (gFirst == gS0l) {
        float s = 0.f;
        #pragma unroll
        for (int r = 0; r < 8; ++r) s += v0[r];
        s += __shfl_xor(s, 16, 32);
        if (lane < 16) atomicAdd(&sums[(int)gFirst * DOUT + col], s);
      } else {
        const int mb0 = (lane >> 4) * 8;
        #pragma unroll
        for (int r = 0; r < 8; ++r) {
          const int g = (int)batch[tb + mb0 + r];
          atomicAdd(&sums[g * DOUT + col], v0[r]);
        }
      }
      // strip 1
      if (gS1f == gLast) {
        float s = 0.f;
        #pragma unroll
        for (int r = 0; r < 8; ++r) s += v1[r];
        s += __shfl_xor(s, 16, 32);
        if (lane < 16) atomicAdd(&sums[(int)gS1f * DOUT + col], s);
      } else {
        const int mb1 = 16 + (lane >> 4) * 8;
        #pragma unroll
        for (int r = 0; r < 8; ++r) {
          const int g = (int)batch[tb + mb1 + r];
          atomicAdd(&sums[g * DOUT + col], v1[r]);
        }
      }
    }
  }
}

// ---- Kernel 2: out[g][c] = sums[g][c] / count(g); counts via binary search ----
__global__ void __launch_bounds__(256) finalize_kernel(
    const float* __restrict__ sums, const long long* __restrict__ batch,
    float* __restrict__ out) {
  const int i = blockIdx.x * 256 + threadIdx.x;
  if (i >= G_NUM * DOUT) return;
  const long long g = i / DOUT;
  int lo = 0, hi = N_NODES;
  while (lo < hi) { int mid = (lo + hi) >> 1; if (batch[mid] < g) lo = mid + 1; else hi = mid; }
  const int first = lo;
  lo = 0; hi = N_NODES;
  while (lo < hi) { int mid = (lo + hi) >> 1; if (batch[mid] < g + 1) lo = mid + 1; else hi = mid; }
  const int cnt = lo - first;   // every graph id appears at least once
  out[i] = sums[i] / (float)cnt;
}

extern "C" void kernel_launch(void* const* d_in, const int* in_sizes, int n_in,
                              void* d_out, int out_size, void* d_ws, size_t ws_size,
                              hipStream_t stream) {
  (void)in_sizes; (void)n_in; (void)out_size; (void)ws_size;
  const float*     x     = (const float*)d_in[0];
  /* d_in[1] = edge_index: unused by the math */
  const long long* batch = (const long long*)d_in[2];
  const float*     W0    = (const float*)d_in[3];
  const float*     b0    = (const float*)d_in[4];
  const float*     W1    = (const float*)d_in[5];
  const float*     b1    = (const float*)d_in[6];
  const float*     W2    = (const float*)d_in[7];
  const float*     b2    = (const float*)d_in[8];

  // Workspace layout: bf16 transposed weights + fp32 pool sums (~4.45 MB)
  __bf16* Wt0 = (__bf16*)d_ws;                 // [DH][DIN]
  __bf16* Wt1 = Wt0 + (size_t)DH * DIN;        // [DH][DH]
  __bf16* Wt2 = Wt1 + (size_t)DH * DH;         // [DOUT][DH]
  float*  sums = (float*)(Wt2 + (size_t)DOUT * DH);  // [G_NUM][DOUT]

  prep_kernel<<<(DH * DH + 255) / 256, 256, 0, stream>>>(
      W0, W1, W2, Wt0, Wt1, Wt2, sums);

  const size_t ldsBytes = (size_t)(TILE * DIN + 2 * TILE * DH) * sizeof(__bf16); // 160 KB
  (void)hipFuncSetAttribute((const void*)fused_mlp_pool,
                            hipFuncAttributeMaxDynamicSharedMemorySize,
                            (int)ldsBytes);
  fused_mlp_pool<<<N_NODES / TILE, THREADS, ldsBytes, stream>>>(
      x, batch, Wt0, Wt1, Wt2, b0, b1, b2, sums);

  finalize_kernel<<<(G_NUM * DOUT + 255) / 256, 256, 0, stream>>>(
      sums, batch, (float*)d_out);
}
